// GraphNeuralNetworkNBF_70128226009225
// MI455X (gfx1250) — matrix operand order repaired and checked by reading, not compiled, not run on hardware
//
#include <hip/hip_runtime.h>

typedef __bf16 bf16;
typedef __bf16 v16bf __attribute__((ext_vector_type(16)));
typedef float  v8f   __attribute__((ext_vector_type(8)));
typedef unsigned int u32x4 __attribute__((ext_vector_type(4)));

union V16 { v16bf v; u32x4 u[2]; };

// ------------------------------------------------------------------
// CDNA5 async global->LDS helpers (ASYNCcnt-tracked data movers)
// ------------------------------------------------------------------
__device__ __forceinline__ void async_copy_b128(unsigned lds_off, const void* g) {
  asm volatile("global_load_async_to_lds_b128 %0, %1, off"
               :: "v"(lds_off), "v"((unsigned long long)(size_t)g)
               : "memory");
}
__device__ __forceinline__ void wait_async0() {
  asm volatile("s_wait_asynccnt 0" ::: "memory");
}

// ------------------------------------------------------------------
// Small utility kernels
// ------------------------------------------------------------------
__global__ void k_f32_to_bf16(const float* __restrict__ in, bf16* __restrict__ out, long n) {
  long i = (long)blockIdx.x * blockDim.x + threadIdx.x;
  if (i < n) out[i] = (bf16)in[i];
}

__global__ void k_embed(const int* __restrict__ ids, const float* __restrict__ tab,
                        bf16* __restrict__ x, int D) {
  int n = blockIdx.x, d = threadIdx.x;
  x[(long)n * D + d] = (bf16)tab[(long)ids[n] * D + d];
}

__global__ void k_hist(const int* __restrict__ et, int* __restrict__ cnt, int E) {
  int e = blockIdx.x * blockDim.x + threadIdx.x;
  if (e < E) {
    int t = et[e] - 1;
    if (t >= 0 && t < 4) atomicAdd(&cnt[t], 1);
  }
}

__global__ void k_offsets(const int* __restrict__ cnt, int* __restrict__ off) {
  if (threadIdx.x == 0 && blockIdx.x == 0) {
    int acc = 0; off[0] = 0;
    for (int t = 0; t < 4; ++t) {
      acc += ((cnt[t] + 15) / 16) * 16;   // pad each type bucket to 16-edge tiles
      off[t + 1] = acc;
    }
  }
}

__global__ void k_scatter(const int* __restrict__ et, int* __restrict__ cur,
                          const int* __restrict__ off, int* __restrict__ sorted, int E) {
  int e = blockIdx.x * blockDim.x + threadIdx.x;
  if (e < E) {
    int t = et[e] - 1;
    if (t >= 0 && t < 4) {
      int pos = atomicAdd(&cur[t], 1);
      sorted[off[t] + pos] = e;
    }
  }
}

// ------------------------------------------------------------------
// Edge-message WMMA kernel.
// A = 16 gathered edge-feature rows (K = KTOT) staged into LDS with
// async global->LDS movers; B = edge_W[type] rows streamed from L2;
// C accumulated f32, scattered into m[tgt] with global f32 atomics.
// Block = 256 threads = 8 waves; wave owns NTW = DOUT/128 column tiles.
// ------------------------------------------------------------------
template <int KTOT, int DOUT, bool HASB>
__global__ __launch_bounds__(256)
void k_msg(const bf16* __restrict__ sA, const bf16* __restrict__ sB,
           const int* __restrict__ sorted, const int* __restrict__ off,
           const int* __restrict__ srcA, const int* __restrict__ tgtA,
           const bf16* __restrict__ W, float* __restrict__ m) {
  constexpr int NTW  = DOUT / 128;
  constexpr int HALF = KTOT / 2;
  constexpr int DA   = 256;
  constexpr int LSTR = KTOT + 32;        // +32 bf16 pad -> 16 rows hit distinct LDS banks
  __shared__ bf16 es[16][LSTR];
  __shared__ int  ltgt[16];

  const int tile = blockIdx.x;
  const int base = tile * 16;
  if (base >= off[4]) return;            // dead padding tile
  int t = 0;
#pragma unroll
  for (int i = 0; i < 4; ++i)
    if (base >= off[i] && base < off[i + 1]) t = i;

  const int lane = threadIdx.x & 31;
  const int wave = threadIdx.x >> 5;

  // ---- stage 16 edge rows into LDS (2 rows per wave, async b128 per lane) ----
  for (int r = wave * 2; r < wave * 2 + 2; ++r) {
    int e = sorted[base + r];            // wave-uniform -> EXEC stays full
    if (lane == 0) ltgt[r] = (e >= 0) ? tgtA[e] : -1;
    unsigned esr = (unsigned)(size_t)&es[r][0];
    if (e >= 0) {
      int sn = srcA[e], tn = tgtA[e];
      async_copy_b128(esr + lane * 16,            sA + (long)sn * DA + lane * 8);
      async_copy_b128(esr + HALF * 2 + lane * 16, sA + (long)tn * DA + lane * 8);
      if (HASB) {
        async_copy_b128(esr + DA * 2 + lane * 16,          sB + (long)sn * DA + lane * 8);
        async_copy_b128(esr + (HALF + DA) * 2 + lane * 16, sB + (long)tn * DA + lane * 8);
      }
    } else {
      u32x4 z = {0u, 0u, 0u, 0u};
      for (int q = lane; q < KTOT / 8; q += 32) ((u32x4*)&es[r][0])[q] = z;
    }
  }
  // warm GL2 with this wave's weight panel (reused by every tile of this type)
#pragma unroll
  for (int j = 0; j < NTW; ++j) {
    int ncol = (wave * NTW + j) * 16 + (lane & 15);
    __builtin_prefetch(W + ((long)t * DOUT + ncol) * KTOT, 0, 1);
  }
  wait_async0();                         // drain this wave's ASYNCcnt before barrier
  __syncthreads();

  const int arow = lane & 15;            // A row / B-C column within tile
  const int ksel = lane >> 4;
  v8f acc[NTW] = {};

  for (int k0 = 0; k0 < KTOT; k0 += 32) {
    // A fragment: lane<16 holds K {k0..k0+7, k0+16..k0+23}; lane>=16 shifted by 8
    V16 a;
    a.u[0] = *(const u32x4*)&es[arow][k0 + 8 * ksel];
    a.u[1] = *(const u32x4*)&es[arow][k0 + 16 + 8 * ksel];
#pragma unroll
    for (int j = 0; j < NTW; ++j) {
      int ncol = (wave * NTW + j) * 16 + arow;
      const bf16* wr = W + ((long)t * DOUT + ncol) * KTOT + k0 + 8 * ksel;
      V16 b;
      b.u[0] = ((const u32x4*)wr)[0];
      b.u[1] = ((const u32x4*)wr)[2];
      acc[j] = __builtin_amdgcn_wmma_f32_16x16x32_bf16(
          false, a.v, false, b.v, (short)0, acc[j], false, false);
    }
  }

  // ---- scatter-add: C layout = VGPR r -> row (r + 8*ksel), col (lane&15) ----
#pragma unroll
  for (int j = 0; j < NTW; ++j) {
    int col = (wave * NTW + j) * 16 + arow;
#pragma unroll
    for (int r = 0; r < 8; ++r) {
      int tg = ltgt[r + 8 * ksel];
      if (tg >= 0) atomicAdd(&m[(long)tg * DOUT + col], acc[j][r]);
    }
  }
}

// ------------------------------------------------------------------
// gelu(exact) + LayerNorm: one wave32 per row, shfl_xor reductions.
// ------------------------------------------------------------------
template <int D>
__global__ __launch_bounds__(256)
void k_pointwise(const float* __restrict__ m, const float* __restrict__ g,
                 const float* __restrict__ bta, bf16* __restrict__ h, int N) {
  constexpr int V = D / 32;
  const int lane = threadIdx.x & 31, wave = threadIdx.x >> 5;
  const int row = blockIdx.x * 8 + wave;
  if (row >= N) return;
  const float* mr = m + (long)row * D;
  float v[V];
  float sum = 0.f;
#pragma unroll
  for (int i = 0; i < V; ++i) {
    float x = mr[lane + 32 * i];
    x = 0.5f * x * (1.0f + erff(x * 0.7071067811865475f));   // exact gelu
    v[i] = x; sum += x;
  }
#pragma unroll
  for (int o = 16; o > 0; o >>= 1) sum += __shfl_xor(sum, o, 32);
  float mu = sum * (1.0f / D);
  float var = 0.f;
#pragma unroll
  for (int i = 0; i < V; ++i) { float d = v[i] - mu; var += d * d; }
#pragma unroll
  for (int o = 16; o > 0; o >>= 1) var += __shfl_xor(var, o, 32);
  var *= (1.0f / D);
  float inv = rsqrtf(var + 1e-5f);
  bf16* hr = h + (long)row * D;
#pragma unroll
  for (int i = 0; i < V; ++i) {
    int d = lane + 32 * i;
    hr[d] = (bf16)((v[i] - mu) * inv * g[d] + bta[d]);
  }
}

// ------------------------------------------------------------------
// Dense: out = tanh(h @ W^T + b).  DOUT = 256 always; K = 256 or 512.
// 16-node tile per block (N % 16 == 0), 8 waves x 2 column tiles.
// ------------------------------------------------------------------
template <int K>
__global__ __launch_bounds__(256)
void k_dense(const bf16* __restrict__ h, const bf16* __restrict__ W,
             const float* __restrict__ bias, bf16* __restrict__ ob,
             float* __restrict__ of) {
  constexpr int DOUT = 256;
  constexpr int NTW  = 2;
  const int lane = threadIdx.x & 31, wave = threadIdx.x >> 5;
  const int arow = lane & 15, ksel = lane >> 4;
  const int tile = blockIdx.x;
  const bf16* ha = h + (long)(tile * 16 + arow) * K;
  v8f acc[NTW] = {};

#pragma unroll
  for (int j = 0; j < NTW; ++j) {
    int ncol = (wave * NTW + j) * 16 + arow;
    __builtin_prefetch(W + (long)ncol * K, 0, 1);
  }

  for (int k0 = 0; k0 < K; k0 += 32) {
    V16 a;
    a.u[0] = ((const u32x4*)(ha + k0 + 8 * ksel))[0];
    a.u[1] = ((const u32x4*)(ha + k0 + 16 + 8 * ksel))[0];
#pragma unroll
    for (int j = 0; j < NTW; ++j) {
      int ncol = (wave * NTW + j) * 16 + arow;
      const bf16* wr = W + (long)ncol * K + k0 + 8 * ksel;
      V16 b;
      b.u[0] = ((const u32x4*)wr)[0];
      b.u[1] = ((const u32x4*)wr)[2];
      acc[j] = __builtin_amdgcn_wmma_f32_16x16x32_bf16(
          false, a.v, false, b.v, (short)0, acc[j], false, false);
    }
  }

#pragma unroll
  for (int j = 0; j < NTW; ++j) {
    int col = (wave * NTW + j) * 16 + arow;
#pragma unroll
    for (int r = 0; r < 8; ++r) {
      int node = tile * 16 + r + 8 * ksel;
      float val = tanhf(acc[j][r] + bias[col]);
      if (ob) ob[(long)node * DOUT + col] = (bf16)val;
      if (of) of[(long)node * DOUT + col] = val;
    }
  }
}

// ------------------------------------------------------------------
// Host orchestration
// ------------------------------------------------------------------
extern "C" void kernel_launch(void* const* d_in, const int* in_sizes, int n_in,
                              void* d_out, int out_size, void* d_ws, size_t ws_size,
                              hipStream_t stream) {
  const int*   ids  = (const int*)d_in[0];
  const int*   te   = (const int*)d_in[1];
  const float* tab  = (const float*)d_in[2];
  const float* eWi  = (const float*)d_in[3];   // [6,4,256,512]
  const float* lnsi = (const float*)d_in[4];   // [6,256]
  const float* lnbi = (const float*)d_in[5];
  const float* dWi  = (const float*)d_in[6];   // [6,256,256]
  const float* dbi  = (const float*)d_in[7];
  const float* eWo  = (const float*)d_in[8];   // [2,4,512,1024]
  const float* lnso = (const float*)d_in[9];   // [2,512]
  const float* lnbo = (const float*)d_in[10];
  const float* dWo  = (const float*)d_in[11];  // [2,256,512]
  const float* dbo  = (const float*)d_in[12];

  const int N = in_sizes[0];
  const int E = in_sizes[1] / 3;
  const int D = 256;

  const int* etA  = te;
  const int* srcA = te + E;
  const int* tgtA = te + 2 * E;

  char* p = (char*)d_ws;
  auto carve = [&](long bytes) -> void* {
    void* r = (void*)p;
    p += (bytes + 255) & ~(long)255;
    return r;
  };

  const long neWi = 6L * 4 * D * (2 * D);
  const long neWo = 2L * 4 * (2 * D) * (4 * D);
  const long ndWi = 6L * D * D;
  const long ndWo = 2L * D * (2 * D);
  bf16* beWi = (bf16*)carve(neWi * 2);
  bf16* beWo = (bf16*)carve(neWo * 2);
  bf16* bdWi = (bf16*)carve(ndWi * 2);
  bf16* bdWo = (bf16*)carve(ndWo * 2);
  bf16*  x0  = (bf16*)carve((long)N * D * 2);
  bf16*  x1  = (bf16*)carve((long)N * D * 2);
  bf16*  x2  = (bf16*)carve((long)N * D * 2);
  float* m   = (float*)carve((long)N * 2 * D * 4);
  bf16*  h   = (bf16*)carve((long)N * 2 * D * 2);
  const int sortedCap = ((E + 64 + 15) / 16) * 16;
  int* sorted = (int*)carve((long)sortedCap * 4);
  int* cnt    = (int*)carve(16);
  int* cur    = (int*)carve(16);
  int* off    = (int*)carve(32);

  // bf16 weight copies (tiny vs 192MB L2; state in bf16 halves gather traffic)
  k_f32_to_bf16<<<(int)((neWi + 255) / 256), 256, 0, stream>>>(eWi, beWi, neWi);
  k_f32_to_bf16<<<(int)((neWo + 255) / 256), 256, 0, stream>>>(eWo, beWo, neWo);
  k_f32_to_bf16<<<(int)((ndWi + 255) / 256), 256, 0, stream>>>(dWi, bdWi, ndWi);
  k_f32_to_bf16<<<(int)((ndWo + 255) / 256), 256, 0, stream>>>(dWo, bdWo, ndWo);

  // embedding gather -> x0 (bf16)
  k_embed<<<N, 256, 0, stream>>>(ids, tab, x0, D);

  // bin edges by type once (types are layer-invariant)
  hipMemsetAsync(cnt, 0, 16, stream);
  hipMemsetAsync(cur, 0, 16, stream);
  hipMemsetAsync(sorted, 0xFF, (long)sortedCap * 4, stream);   // -1 sentinels
  k_hist<<<(E + 255) / 256, 256, 0, stream>>>(etA, cnt, E);
  k_offsets<<<1, 1, 0, stream>>>(cnt, off);
  k_scatter<<<(E + 255) / 256, 256, 0, stream>>>(etA, cur, off, sorted, E);

  const int tiles  = sortedCap / 16;
  const int ntileN = N / 16;     // N=100000 -> 6250

  // buffer schedule (block input must survive for residual concat)
  bf16* innerIn[2][3]  = {{x0, x1, x2}, {x2, x1, x0}};
  bf16* innerOut[2][3] = {{x1, x2, x1}, {x1, x0, x1}};
  bf16* blockIn[2]     = {x0, x2};

  for (int b = 0; b < 2; ++b) {
    for (int i = 0; i < 3; ++i) {
      const int L = 3 * b + i;
      hipMemsetAsync(m, 0, (long)N * D * 4, stream);
      k_msg<512, 256, false><<<tiles, 256, 0, stream>>>(
          innerIn[b][i], nullptr, sorted, off, srcA, tgtA,
          beWi + (long)L * 4 * D * 2 * D, m);
      k_pointwise<256><<<N / 8, 256, 0, stream>>>(
          m, lnsi + (long)L * D, lnbi + (long)L * D, h, N);
      k_dense<256><<<ntileN, 256, 0, stream>>>(
          h, bdWi + (long)L * D * D, dbi + (long)L * D, innerOut[b][i], nullptr);
    }
    // out layer on concat [blockIn, x1]
    hipMemsetAsync(m, 0, (long)N * 2 * D * 4, stream);
    k_msg<1024, 512, true><<<tiles, 256, 0, stream>>>(
        blockIn[b], x1, sorted, off, srcA, tgtA,
        beWo + (long)b * 4 * (2 * D) * (4 * D), m);
    k_pointwise<512><<<N / 8, 256, 0, stream>>>(
        m, lnso + (long)b * 2 * D, lnbo + (long)b * 2 * D, h, N);
    if (b == 0) {
      k_dense<512><<<ntileN, 256, 0, stream>>>(
          h, bdWo + (long)b * D * 2 * D, dbo + (long)b * D, x2, nullptr);
    } else {
      k_dense<512><<<ntileN, 256, 0, stream>>>(
          h, bdWo + (long)b * D * 2 * D, dbo + (long)b * D, nullptr, (float*)d_out);
    }
  }
}
